// Block_24455543783711
// MI455X (gfx1250) — matrix operand order repaired and checked by reading, not compiled, not run on hardware
//
#include <hip/hip_runtime.h>
#include <stdint.h>

#define TT   1024
#define CC   768
#define HH   12
#define HDIM 64
#define MTOT 4096   // B*T

typedef __attribute__((ext_vector_type(16))) _Float16 v16h;
typedef __attribute__((ext_vector_type(8)))  float    v8f;
typedef __attribute__((ext_vector_type(4)))  uint32_t u32x4;
typedef __attribute__((ext_vector_type(8)))  uint32_t u32x8;

union Frag16 { v16h v; uint4 q[2]; };

// LDS row pitch for TDM-staged B tiles: 64 f16 + 8 f16 skew (16B) -> 144B rows,
// keeps ds_load_b128 16B-aligned and reduces lane=row bank conflicts.
#define BPITCH 72

// A-matrix 16x32 f16 fragment (ISA 7.12.2): lane<16 -> M=lane, K elems {0..7,16..23};
// lane>=16 -> M=lane-16, K elems {8..15,24..31}. Two contiguous 16B loads per lane.
__device__ inline v16h load_a_16x32(const _Float16* __restrict__ A, int lda, int m0, int k0, int lane) {
  int lo = lane & 15, hi = lane >> 4;
  const _Float16* p = A + (size_t)(m0 + lo) * lda + k0 + 8 * hi;
  Frag16 f;
  f.q[0] = *(const uint4*)(p);
  f.q[1] = *(const uint4*)(p + 16);
  return f.v;
}

// B-matrix 32x16 f16 fragment from global: lane -> N=lane%16, K elems {8h..8h+7, 16+8h..16+8h+7} (h=lane/16).
__device__ inline v16h load_b_32x16(const _Float16* __restrict__ Bm, int ldb, int k0, int n0, int lane) {
  int lo = lane & 15, hi = lane >> 4;
  const _Float16* p = Bm + (size_t)(k0 + 8 * hi) * ldb + n0 + lo;
  v16h b;
#pragma unroll
  for (int s = 0; s < 8; ++s) { b[s] = p[(size_t)s * ldb]; b[8 + s] = p[(size_t)(16 + s) * ldb]; }
  return b;
}

// Tensor Data Mover: DMA a 32(row) x 64(col) f16 tile (row stride = N elements)
// from global into LDS at lds_off, with a 16B pad after every 128B row.
// D# per CDNA5 ISA ch.8: group0 = {count/lds_addr/global_addr/type},
// group1 = {data_size, pad, tensor_dim0/1, tile_dim0/1, tensor_dim0_stride}.
__device__ inline void tdm_load_b32x64(const _Float16* g, int N, uint32_t lds_off) {
  uint64_t ga = (uint64_t)(uintptr_t)g;
  u32x4 g0;
  g0[0] = 1u;                                               // count=1, user mode
  g0[1] = lds_off;                                          // lds_addr (bytes)
  g0[2] = (uint32_t)ga;                                     // global_addr[31:0]
  g0[3] = ((uint32_t)(ga >> 32) & 0x01FFFFFFu) | (2u << 30); // addr[56:32] | type=2
  u32x8 g1;
  g1[0] = (1u << 16)            // data_size = 1 (2 bytes)
        | (1u << 20)            // pad_enable
        | (4u << 22)            // pad_interval: code 4 = every 32 DWORDs (128B row)
        | (3u << 25);           // pad_amount:   code 3 = 4 DWORDs (16B)
  g1[1] = ((uint32_t)N & 0xFFFFu) << 16;                    // tensor_dim0[15:0]
  g1[2] = (((uint32_t)N >> 16) & 0xFFFFu) | (32u << 16);    // tensor_dim0[31:16] | tensor_dim1[15:0]=32
  g1[3] = (64u << 16);                                      // tensor_dim1[31:16]=0 | tile_dim0=64
  g1[4] = 32u;                                              // tile_dim1=32, tile_dim2=0
  g1[5] = (uint32_t)N;                                      // tensor_dim0_stride[31:0]
  g1[6] = 0u;                                               // stride[47:32]=0, dim1_stride lo=0
  g1[7] = 0u;
  asm volatile("tensor_load_to_lds %0, %1" :: "s"(g0), "s"(g1) : "memory");
}

__device__ inline float gelu_ref(float x) {
  // faithful to reference: 0.5*x*(1+tanh(sqrt(2/pi)*(x*0.044715*x^3)))
  float x4 = x * x * x * x;
  return 0.5f * x * (1.0f + tanhf(0.7978845608028654f * (0.044715f * x4)));
}

__global__ void f2h_k(const float* __restrict__ s, _Float16* __restrict__ d, int n) {
  int i = blockIdx.x * 256 + threadIdx.x;
  if (i < n) d[i] = (_Float16)s[i];
}

// one row (768) per 256-thread block; writes f16 normalized row
__global__ void ln_f16_k(const float* __restrict__ x, const float* __restrict__ g,
                         const float* __restrict__ b, _Float16* __restrict__ o) {
  int row = blockIdx.x;
  int tid = threadIdx.x;
  const float* xr = x + (size_t)row * CC;
  float v0 = xr[tid], v1 = xr[tid + 256], v2 = xr[tid + 512];
  __shared__ float sh[512];
  sh[tid]       = v0 + v1 + v2;
  sh[tid + 256] = v0 * v0 + v1 * v1 + v2 * v2;
  __syncthreads();
  for (int off = 128; off > 0; off >>= 1) {
    if (tid < off) { sh[tid] += sh[tid + off]; sh[tid + 256] += sh[tid + 256 + off]; }
    __syncthreads();
  }
  float mean = sh[0]   * (1.0f / CC);
  float var  = sh[256] * (1.0f / CC) - mean * mean;
  float rs = rsqrtf(var + 1e-5f);
  _Float16* orow = o + (size_t)row * CC;
  orow[tid]       = (_Float16)((v0 - mean) * rs * g[tid]       + b[tid]);
  orow[tid + 256] = (_Float16)((v1 - mean) * rs * g[tid + 256] + b[tid + 256]);
  orow[tid + 512] = (_Float16)((v2 - mean) * rs * g[tid + 512] + b[tid + 512]);
}

// Generic WMMA GEMM, TDM-staged B: each wave computes a 32(M) x 64(N) tile and
// owns a private double-buffered LDS B panel filled by tensor_load_to_lds.
// MODE 0: outF = acc + bias + resid (f32).  MODE 1: outH = gelu(acc + bias) (f16).
template<int MODE>
__global__ void gemm_wmma_k(const _Float16* __restrict__ A, const _Float16* __restrict__ W,
                            const float* __restrict__ bias, const float* __restrict__ resid,
                            float* __restrict__ outF, _Float16* __restrict__ outH,
                            int N, int K) {
  __shared__ _Float16 bsm[4][2][32][BPITCH];   // 4 waves x double buffer x 32x(64+8)
  int lane = threadIdx.x;
  int w    = threadIdx.y;
  int n0 = blockIdx.x * 64;
  int m0 = (blockIdx.y * blockDim.y + w) * 32;

  tdm_load_b32x64(W + n0, N, (uint32_t)(uintptr_t)&bsm[w][0][0][0]);

  v8f acc[2][4] = {};
  int cur = 0;
  for (int k0 = 0; k0 < K; k0 += 32) {
    if (k0 + 32 < K) {
      tdm_load_b32x64(W + (size_t)(k0 + 32) * N + n0, N,
                      (uint32_t)(uintptr_t)&bsm[w][cur ^ 1][0][0]);
      asm volatile("s_wait_tensorcnt 0x1" ::: "memory");   // current buffer ready
    } else {
      asm volatile("s_wait_tensorcnt 0x0" ::: "memory");
    }
    v16h a0 = load_a_16x32(A, K, m0,      k0, lane);
    v16h a1 = load_a_16x32(A, K, m0 + 16, k0, lane);
#pragma unroll
    for (int t = 0; t < 4; ++t) {
      Frag16 f;                                             // B frag: lane%16 = N col, K in A order
#pragma unroll
      for (int s = 0; s < 8; ++s) {
        f.v[s]     = bsm[w][cur][8 * (lane >> 4) + s][16 * t + (lane & 15)];
        f.v[8 + s] = bsm[w][cur][16 + 8 * (lane >> 4) + s][16 * t + (lane & 15)];
      }
      acc[0][t] = __builtin_amdgcn_wmma_f32_16x16x32_f16(false, a0, false, f.v, (short)0, acc[0][t], false, false);
      acc[1][t] = __builtin_amdgcn_wmma_f32_16x16x32_f16(false, a1, false, f.v, (short)0, acc[1][t], false, false);
    }
    cur ^= 1;
  }
  int lo = lane & 15, hi = lane >> 4;
#pragma unroll
  for (int s = 0; s < 2; ++s)
#pragma unroll
    for (int t = 0; t < 4; ++t) {
      int n = n0 + 16 * t + lo;
      float bv = bias[n];
#pragma unroll
      for (int r = 0; r < 8; ++r) {
        int m = m0 + 16 * s + r + 8 * hi;     // C layout: M = reg + 8*(lane>=16)
        size_t idx = (size_t)m * N + n;
        float v = acc[s][t][r] + bv;
        if (MODE == 0) outF[idx] = v + resid[idx];
        else           outH[idx] = (_Float16)gelu_ref(v);
      }
    }
}

// QKV GEMM (TDM-staged B) with head-scatter epilogue. Per-head columns: [q64|k64|v64].
// Writes Q [B,H,T,64], K^T [B,H,64,T] (transposed store), V [B,H,T,64], all f16.
__global__ void gemm_qkv_k(const _Float16* __restrict__ A, const _Float16* __restrict__ W,
                           const float* __restrict__ bias,
                           _Float16* __restrict__ qb, _Float16* __restrict__ ktb,
                           _Float16* __restrict__ vb) {
  const int K = CC, N = 3 * CC;
  __shared__ _Float16 bsm[4][2][32][BPITCH];
  int lane = threadIdx.x;
  int w    = threadIdx.y;
  int n0 = blockIdx.x * 64;
  int m0 = (blockIdx.y * blockDim.y + w) * 32;

  tdm_load_b32x64(W + n0, N, (uint32_t)(uintptr_t)&bsm[w][0][0][0]);

  v8f acc[2][4] = {};
  int cur = 0;
  for (int k0 = 0; k0 < K; k0 += 32) {
    if (k0 + 32 < K) {
      tdm_load_b32x64(W + (size_t)(k0 + 32) * N + n0, N,
                      (uint32_t)(uintptr_t)&bsm[w][cur ^ 1][0][0]);
      asm volatile("s_wait_tensorcnt 0x1" ::: "memory");
    } else {
      asm volatile("s_wait_tensorcnt 0x0" ::: "memory");
    }
    v16h a0 = load_a_16x32(A, K, m0,      k0, lane);
    v16h a1 = load_a_16x32(A, K, m0 + 16, k0, lane);
#pragma unroll
    for (int t = 0; t < 4; ++t) {
      Frag16 f;
#pragma unroll
      for (int s = 0; s < 8; ++s) {
        f.v[s]     = bsm[w][cur][8 * (lane >> 4) + s][16 * t + (lane & 15)];
        f.v[8 + s] = bsm[w][cur][16 + 8 * (lane >> 4) + s][16 * t + (lane & 15)];
      }
      acc[0][t] = __builtin_amdgcn_wmma_f32_16x16x32_f16(false, a0, false, f.v, (short)0, acc[0][t], false, false);
      acc[1][t] = __builtin_amdgcn_wmma_f32_16x16x32_f16(false, a1, false, f.v, (short)0, acc[1][t], false, false);
    }
    cur ^= 1;
  }
  int lo = lane & 15, hi = lane >> 4;
  int head = n0 / 192;
  int sect = (n0 % 192) / 64;   // 0=q 1=k 2=v (64-wide tile never straddles)
#pragma unroll
  for (int s = 0; s < 2; ++s)
#pragma unroll
    for (int t = 0; t < 4; ++t) {
      int d = 16 * t + lo;
      float bv = bias[n0 + 16 * t + lo];
#pragma unroll
      for (int r = 0; r < 8; ++r) {
        int m = m0 + 16 * s + r + 8 * hi;
        int bb = m >> 10;           // batch
        int tt = m & 1023;          // token
        int bh = bb * HH + head;
        _Float16 hv = (_Float16)(acc[s][t][r] + bv);
        if (sect == 0)      qb [((size_t)bh * TT + tt) * HDIM + d] = hv;
        else if (sect == 1) ktb[((size_t)bh * HDIM + d) * TT + tt] = hv;   // transposed
        else                vb [((size_t)bh * TT + tt) * HDIM + d] = hv;
      }
    }
}

// Flash attention: one wave per 16-query tile per (b,h); streams 32 keys/step.
__global__ void attn_k(const _Float16* __restrict__ qb, const _Float16* __restrict__ ktb,
                       const _Float16* __restrict__ vb, _Float16* __restrict__ y) {
  int lane = threadIdx.x;
  int w    = threadIdx.y;
  int bh   = blockIdx.y;                 // 0..47
  int t0   = (blockIdx.x * 4 + w) * 16;  // query tile base
  int bb = bh / HH, h = bh % HH;
  const _Float16* Q  = qb  + (size_t)bh * TT * HDIM;
  const _Float16* KT = ktb + (size_t)bh * HDIM * TT;
  const _Float16* V  = vb  + (size_t)bh * TT * HDIM;
  int lo = lane & 15, hi = lane >> 4;

  v16h aq0 = load_a_16x32(Q, HDIM, t0, 0,  lane);   // d = 0..31
  v16h aq1 = load_a_16x32(Q, HDIM, t0, 32, lane);   // d = 32..63

  v8f o[4] = {};
  float mrow[8], lrow[8];
#pragma unroll
  for (int r = 0; r < 8; ++r) { mrow[r] = -3.0e38f; lrow[r] = 0.0f; }

  __shared__ _Float16 plds[4][16][32];   // per-wave P tile (C->A layout bounce)

  for (int k0 = 0; k0 <= t0; k0 += 32) {
    // S = Q * K^T  (two 16-key C tiles, contraction over d in two 32-chunks)
    v8f sA = {}, sB = {};
    {
      v16h b0 = load_b_32x16(KT, TT, 0,  k0, lane);
      v16h b1 = load_b_32x16(KT, TT, 32, k0, lane);
      sA = __builtin_amdgcn_wmma_f32_16x16x32_f16(false, aq0, false, b0, (short)0, sA, false, false);
      sA = __builtin_amdgcn_wmma_f32_16x16x32_f16(false, aq1, false, b1, (short)0, sA, false, false);
      v16h c0 = load_b_32x16(KT, TT, 0,  k0 + 16, lane);
      v16h c1 = load_b_32x16(KT, TT, 32, k0 + 16, lane);
      sB = __builtin_amdgcn_wmma_f32_16x16x32_f16(false, aq0, false, c0, (short)0, sB, false, false);
      sB = __builtin_amdgcn_wmma_f32_16x16x32_f16(false, aq1, false, c1, (short)0, sB, false, false);
    }
    // scale, causal mask, online softmax (row reductions inside 16-lane halves)
#pragma unroll
    for (int r = 0; r < 8; ++r) {
      int qi = t0 + r + 8 * hi;
      float v0 = sA[r] * 0.125f;          // 1/sqrt(64)
      float v1 = sB[r] * 0.125f;
      if (k0 + lo > qi)      v0 = -1.0e30f;
      if (k0 + 16 + lo > qi) v1 = -1.0e30f;
      float mx = fmaxf(v0, v1);
      mx = fmaxf(mx, __shfl_xor(mx, 1));
      mx = fmaxf(mx, __shfl_xor(mx, 2));
      mx = fmaxf(mx, __shfl_xor(mx, 4));
      mx = fmaxf(mx, __shfl_xor(mx, 8));
      float mn  = fmaxf(mrow[r], mx);
      float fac = __expf(mrow[r] - mn);
      mrow[r] = mn;
      float p0 = __expf(v0 - mn);
      float p1 = __expf(v1 - mn);
      float sum = p0 + p1;
      sum += __shfl_xor(sum, 1);
      sum += __shfl_xor(sum, 2);
      sum += __shfl_xor(sum, 4);
      sum += __shfl_xor(sum, 8);
      lrow[r] = lrow[r] * fac + sum;
#pragma unroll
      for (int t = 0; t < 4; ++t) o[t][r] = o[t][r] * fac;
      plds[w][r + 8 * hi][lo]      = (_Float16)p0;   // row-major P tile in LDS
      plds[w][r + 8 * hi][16 + lo] = (_Float16)p1;
    }
    // re-read P as a 16x32 A fragment
    Frag16 pf;
    pf.q[0] = *(const uint4*)&plds[w][lo][8 * hi];
    pf.q[1] = *(const uint4*)&plds[w][lo][8 * hi + 16];
    // O += P * V  (4 d-tiles of 16)
#pragma unroll
    for (int t = 0; t < 4; ++t) {
      v16h bv_ = load_b_32x16(V, HDIM, k0, 16 * t, lane);
      o[t] = __builtin_amdgcn_wmma_f32_16x16x32_f16(false, pf.v, false, bv_, (short)0, o[t], false, false);
    }
  }
  // normalize and write y back to [B,T,C] token-major (f16, feeds O-proj GEMM)
#pragma unroll
  for (int t = 0; t < 4; ++t)
#pragma unroll
    for (int r = 0; r < 8; ++r) {
      int tt = t0 + r + 8 * hi;
      float val = o[t][r] / lrow[r];
      y[(size_t)(bb * TT + tt) * CC + h * HDIM + 16 * t + lo] = (_Float16)val;
    }
}

extern "C" void kernel_launch(void* const* d_in, const int* in_sizes, int n_in,
                              void* d_out, int out_size, void* d_ws, size_t ws_size,
                              hipStream_t stream) {
  (void)in_sizes; (void)n_in; (void)out_size; (void)ws_size;
  const float* x     = (const float*)d_in[0];
  const float* g1    = (const float*)d_in[1];
  const float* b1    = (const float*)d_in[2];
  const float* Wqkv  = (const float*)d_in[3];
  const float* bqkv  = (const float*)d_in[4];
  const float* Wo    = (const float*)d_in[5];
  const float* bo    = (const float*)d_in[6];
  const float* g2    = (const float*)d_in[7];
  const float* b2    = (const float*)d_in[8];
  const float* Wfc   = (const float*)d_in[9];
  const float* bfc   = (const float*)d_in[10];
  const float* Wproj = (const float*)d_in[11];
  const float* bproj = (const float*)d_in[12];
  float* out = (float*)d_out;

  char* ws = (char*)d_ws;
  size_t off = 0;
  auto alloc = [&](size_t bytes) -> void* {
    void* p = ws + off;
    off += (bytes + 255) & ~(size_t)255;
    return p;
  };
  _Float16* ln1h   = (_Float16*)alloc((size_t)MTOT * CC * 2);
  _Float16* wqkvh  = (_Float16*)alloc((size_t)CC * 3 * CC * 2);
  _Float16* woh    = (_Float16*)alloc((size_t)CC * CC * 2);
  _Float16* wfch   = (_Float16*)alloc((size_t)CC * 4 * CC * 2);
  _Float16* wprojh = (_Float16*)alloc((size_t)4 * CC * CC * 2);
  _Float16* qh     = (_Float16*)alloc((size_t)MTOT * CC * 2);
  _Float16* kth    = (_Float16*)alloc((size_t)MTOT * CC * 2);
  _Float16* vh     = (_Float16*)alloc((size_t)MTOT * CC * 2);
  _Float16* yh     = (_Float16*)alloc((size_t)MTOT * CC * 2);
  float*    x2     = (float*)   alloc((size_t)MTOT * CC * 4);
  _Float16* ln2h   = (_Float16*)alloc((size_t)MTOT * CC * 2);
  _Float16* hh     = (_Float16*)alloc((size_t)MTOT * 4 * CC * 2);

  dim3 blk(32, 4);

  // weight fp32 -> f16
  {
    int n;
    n = CC * 3 * CC;  f2h_k<<<(n + 255) / 256, 256, 0, stream>>>(Wqkv,  wqkvh,  n);
    n = CC * CC;      f2h_k<<<(n + 255) / 256, 256, 0, stream>>>(Wo,    woh,    n);
    n = CC * 4 * CC;  f2h_k<<<(n + 255) / 256, 256, 0, stream>>>(Wfc,   wfch,   n);
    n = 4 * CC * CC;  f2h_k<<<(n + 255) / 256, 256, 0, stream>>>(Wproj, wprojh, n);
  }

  // LN1
  ln_f16_k<<<MTOT, 256, 0, stream>>>(x, g1, b1, ln1h);

  // QKV GEMM (4096x2304, K=768) + head scatter
  gemm_qkv_k<<<dim3(3 * CC / 64, MTOT / 128), blk, 0, stream>>>(ln1h, wqkvh, bqkv, qh, kth, vh);

  // attention: grid (T/64, B*H), 4 q-tiles of 16 per block
  attn_k<<<dim3(TT / 64, 4 * HH), blk, 0, stream>>>(qh, kth, vh, yh);

  // O-proj + residual -> x2 (f32)
  gemm_wmma_k<0><<<dim3(CC / 64, MTOT / 128), blk, 0, stream>>>(yh, woh, bo, x, x2, nullptr, CC, CC);

  // LN2
  ln_f16_k<<<MTOT, 256, 0, stream>>>(x2, g2, b2, ln2h);

  // FC + gelu -> hh (f16), N=3072
  gemm_wmma_k<1><<<dim3(4 * CC / 64, MTOT / 128), blk, 0, stream>>>(ln2h, wfch, bfc, nullptr, nullptr, hh, 4 * CC, CC);

  // Proj + residual -> out (f32), K=3072
  gemm_wmma_k<0><<<dim3(CC / 64, MTOT / 128), blk, 0, stream>>>(hh, wprojh, bproj, x2, out, nullptr, CC, 4 * CC);
}
